// HandwritingModel_13271448945279
// MI455X (gfx1250) — matrix operand order repaired and checked by reading, not compiled, not run on hardware
//
#include <hip/hip_runtime.h>
#include <hip/hip_bf16.h>
#include <math.h>

// ---------------------------------------------------------------------------
// Graves handwriting model forward pass for MI455X (gfx1250, wave32, WMMA).
//   T=1024 sequential LSTM steps, B=128, H=512, attention over U=64 chars,
//   fused mixture-density NLL -> single scalar.
// 16 workgroups (one per WGP), each owns 32 hidden units (=128 gate rows).
// Cell state lives in VGPRs for the whole sequence. One grid barrier / step
// (atomic counter + s_cluster_barrier). GEMMs (gates, attention projection,
// mixture head) all use v_wmma_f32_16x16x32_bf16. W_hh / W_att / W_ih weight
// slices are LDS-resident (304.6 KB of the 320 KB/WG CDNA5 LDS), preloaded
// and per-step staged with async global->LDS loads (ASYNCcnt).
// ---------------------------------------------------------------------------

#define TPB 256
#define NWG 16

constexpr int T_STEPS = 1024;
constexpr int BATCH   = 128;
constexpr int NH      = 512;
constexpr int NA      = 10;
constexpr int NG      = 20;
constexpr int NC      = 80;
constexpr int XD      = 96;    // x = [stroke(3) | w(80)] padded to 96
constexpr int GSL     = 128;   // gate-slice rows per WG (4 gates x 32 hidden)
constexpr int MIXIN   = 608;   // 512 + 80 padded to 608
constexpr int MIXO    = 128;   // 121 mixture outputs padded to 128

typedef __attribute__((ext_vector_type(16))) __bf16 v16bf;
typedef __attribute__((ext_vector_type(8)))  __bf16 v8bf;
typedef __attribute__((ext_vector_type(8)))  float  v8f;

// ---- workspace layout (bytes) ----
constexpr size_t OFF_HDR  = 0;                                     // barrier + accums (256 B)
constexpr size_t OFF_WHH  = 256;                                   // [NWG][128][512] bf16
constexpr size_t OFF_WIH  = OFF_WHH + (size_t)NWG*GSL*NH*2;        // [NWG][128][96]  bf16
constexpr size_t OFF_WATT = OFF_WIH + (size_t)NWG*GSL*XD*2;        // [32][512]       bf16
constexpr size_t OFF_WMIX = OFF_WATT + (size_t)32*NH*2;            // [128][608]      bf16
constexpr size_t OFF_H    = OFF_WMIX + (size_t)MIXO*MIXIN*2;       // [2][128][512]   bf16

// ---- dynamic LDS layout (bytes): 304,640 <= 320 KB/WG (ISA 3.1) ----
constexpr int SMEM_BYTES =
    GSL*NH*2      // whh_lds   (128 KB, resident LSTM recurrent weights)
  + GSL*XD*2      // wih_lds
  + 32*NH*2       // watt_lds  (attention projection weights)
  + BATCH*XD*2    // xbuf (w-state lives in cols 3..82 as bf16)
  + BATCH*32*4    // att_lds (30 padded to 32)
  + BATCH*NC*4    // w_state f32
  + BATCH*NA*4    // k_lds
  + GSL*4         // bias_lds
  + 16*MIXIN*2    // hwbuf
  + 16*MIXO*4     // y_lds
  + TPB*4;        // reduction

// ---------------------------------------------------------------------------
// WMMA operand loaders, following CDNA5 ISA 7.12.2 VGPR layouts (wave32).
// A (16x32 bf16): lane m=l&15,g=l>>4 holds K segments [8g,8g+8) and [16+8g,+8)
// B (32x16 bf16): lane n=l&15,g=l>>4 holds K segments [8g,8g+8) and [16+8g,+8) of col n
// ---------------------------------------------------------------------------
__device__ __forceinline__ v16bf load_a16(const __bf16* tile, int stride, int lane) {
  const int m = lane & 15, g = lane >> 4;
  const __bf16* p = tile + m * stride + 8 * g;
  v8bf lo = *(const v8bf*)(p);
  v8bf hi = *(const v8bf*)(p + 16);
  v16bf a;
#pragma unroll
  for (int i = 0; i < 8; ++i) { a[i] = lo[i]; a[i + 8] = hi[i]; }
  return a;
}
__device__ __forceinline__ v16bf load_b16(const __bf16* tile, int stride, int lane) {
  const int nn = lane & 15, g = lane >> 4;
  const __bf16* p = tile + nn * stride + 8 * g;
  v8bf lo = *(const v8bf*)(p);
  v8bf hi = *(const v8bf*)(p + 16);
  v16bf b;
#pragma unroll
  for (int i = 0; i < 8; ++i) { b[i] = lo[i]; b[i + 8] = hi[i]; }
  return b;
}

// Async global->LDS copy of one 16-byte chunk (CDNA5 async-DMA path,
// tracked by ASYNCcnt).  dst: wave-relative LDS byte address; src: global.
__device__ __forceinline__ void async_copy_b128(unsigned lds_addr, const void* gsrc) {
  asm volatile("global_load_async_to_lds_b128 %0, %1, off"
               :: "v"(lds_addr), "v"((unsigned long long)gsrc) : "memory");
}
__device__ __forceinline__ void async_wait0() {
  asm volatile("s_wait_asynccnt 0" ::: "memory");
}
__device__ __forceinline__ unsigned lds_addr_of(const void* p) {
  return (unsigned)(size_t)p;                  // LDS flat addr: low 32 bits
}

__device__ __forceinline__ float sigf(float x)    { return 1.f / (1.f + expf(-x)); }
__device__ __forceinline__ float logsigf(float x) { return (x > 0.f) ? -log1pf(expf(-x)) : x - log1pf(expf(x)); }

__device__ __forceinline__ float mix_logn(const float* y, int j, float x1, float x2) {
  const float LOG2PI = 1.8378770664093453f;
  float mu1 = y[20 + j], mu2 = y[40 + j], ls1 = y[60 + j], ls2 = y[80 + j];
  float rh  = tanhf(y[100 + j]);
  float i1  = expf(-ls1), i2 = expf(-ls2);
  float d1  = (x1 - mu1) * i1, d2 = (x2 - mu2) * i2;
  float omr = 1.f - rh * rh;
  float Z   = d1 * d1 + d2 * d2 - 2.f * rh * d1 * d2;
  return -Z / (2.f * omr) - LOG2PI - ls1 - ls2 - 0.5f * logf(omr);
}

// ---------------------------------------------------------------------------
// Prep: convert / re-layout weights to bf16 slices, zero barrier/accum/h.
// ---------------------------------------------------------------------------
__global__ void hw_prep(const float* __restrict__ W_ih, const float* __restrict__ W_hh,
                        const float* __restrict__ W_att, const float* __restrict__ W_mix,
                        char* __restrict__ ws)
{
  __bf16* whh  = (__bf16*)(ws + OFF_WHH);
  __bf16* wih  = (__bf16*)(ws + OFF_WIH);
  __bf16* watt = (__bf16*)(ws + OFF_WATT);
  __bf16* wmix = (__bf16*)(ws + OFF_WMIX);
  __bf16* hbf  = (__bf16*)(ws + OFF_H);
  unsigned* hdr = (unsigned*)(ws + OFF_HDR);

  const long long N0 = 64;                            // header dwords (256 B)
  const long long N1 = (long long)NWG * GSL * NH;     // whh
  const long long N2 = (long long)NWG * GSL * XD;     // wih
  const long long N3 = 32LL * NH;                     // watt
  const long long N4 = (long long)MIXO * MIXIN;       // wmix
  const long long N5 = 2LL * BATCH * NH;              // h zero
  const long long TOT = N0 + N1 + N2 + N3 + N4 + N5;

  for (long long i = (long long)blockIdx.x * blockDim.x + threadIdx.x; i < TOT;
       i += (long long)gridDim.x * blockDim.x) {
    long long j = i;
    if (j < N0) { hdr[j] = 0u; continue; }
    j -= N0;
    if (j < N1) {
      int wg = (int)(j / (GSL * NH)); int rem = (int)(j % (GSL * NH));
      int r = rem / NH, k = rem % NH;
      int gr = (r >> 5) * NH + wg * 32 + (r & 31);    // [i|f|g|o] gate row
      whh[j] = (__bf16)W_hh[(long long)gr * NH + k];
      continue;
    }
    j -= N1;
    if (j < N2) {
      int wg = (int)(j / (GSL * XD)); int rem = (int)(j % (GSL * XD));
      int r = rem / XD, k = rem % XD;
      int gr = (r >> 5) * NH + wg * 32 + (r & 31);
      wih[j] = (k < 83) ? (__bf16)W_ih[(long long)gr * 83 + k] : (__bf16)0.f;
      continue;
    }
    j -= N2;
    if (j < N3) {
      int jj = (int)(j / NH), k = (int)(j % NH);
      watt[j] = (jj < 30) ? (__bf16)W_att[(long long)jj * NH + k] : (__bf16)0.f;
      continue;
    }
    j -= N3;
    if (j < N4) {
      int r = (int)(j / MIXIN), k = (int)(j % MIXIN);
      wmix[j] = (r < 121 && k < 592) ? (__bf16)W_mix[(long long)r * 592 + k] : (__bf16)0.f;
      continue;
    }
    j -= N4;
    hbf[j] = (__bf16)0.f;
  }
}

// ---------------------------------------------------------------------------
// Main persistent kernel: 16 WGs, 256 threads (8 waves) each.
// ---------------------------------------------------------------------------
__global__ __launch_bounds__(TPB)
void hw_main(const float* __restrict__ seq_pt, const float* __restrict__ seq_mask,
             const float* __restrict__ tgt,    const int*   __restrict__ cidx,
             const float* __restrict__ c_mask,
             const float* __restrict__ b_ih,   const float* __restrict__ b_hh,
             const float* __restrict__ b_att,  const float* __restrict__ b_mix,
             char* __restrict__ ws)
{
  extern __shared__ char smem[];
  __bf16* whh_lds  = (__bf16*)smem;                       // GSL*NH
  __bf16* wih_lds  = whh_lds + GSL * NH;                  // GSL*XD
  __bf16* watt_lds = wih_lds + GSL * XD;                  // 32*NH
  __bf16* xbuf     = watt_lds + 32 * NH;                  // BATCH*XD
  float*  att_lds  = (float*)(xbuf + BATCH * XD);         // BATCH*32
  float*  w_state  = att_lds + BATCH * 32;                // BATCH*NC
  float*  k_lds    = w_state + BATCH * NC;                // BATCH*NA
  float*  bias_lds = k_lds + BATCH * NA;                  // GSL
  __bf16* hwbuf    = (__bf16*)(bias_lds + GSL);           // 16*MIXIN
  float*  y_lds    = (float*)(hwbuf + 16 * MIXIN);        // 16*MIXO
  float*  red      = y_lds + 16 * MIXO;                   // TPB

  const __bf16* whh_g  = (const __bf16*)(ws + OFF_WHH) + (size_t)blockIdx.x * GSL * NH;
  const __bf16* wih_g  = (const __bf16*)(ws + OFF_WIH) + (size_t)blockIdx.x * GSL * XD;
  const __bf16* watt_g = (const __bf16*)(ws + OFF_WATT);
  const __bf16* wmix   = (const __bf16*)(ws + OFF_WMIX);
  __bf16* hglob = (__bf16*)(ws + OFF_H);
  int*   bar   = (int*)(ws + OFF_HDR);
  float* accum = (float*)(ws + OFF_HDR + 64);

  const int wg = blockIdx.x, tid = threadIdx.x;
  const int wave = tid >> 5, lane = tid & 31;
  const int nn = lane & 15, g = lane >> 4;

  // ---- LDS init: weight slices via async global->LDS DMA, states zeroed ----
  for (int i = tid; i < (GSL * NH) / 8; i += TPB)         // W_hh slice, 128 KB
    async_copy_b128(lds_addr_of(whh_lds + i * 8), whh_g + i * 8);
  for (int i = tid; i < (GSL * XD) / 8; i += TPB)         // W_ih slice
    async_copy_b128(lds_addr_of(wih_lds + i * 8), wih_g + i * 8);
  for (int i = tid; i < (32 * NH) / 8; i += TPB)          // W_att
    async_copy_b128(lds_addr_of(watt_lds + i * 8), watt_g + i * 8);
  for (int i = tid; i < BATCH * XD; i += TPB) xbuf[i] = (__bf16)0.f;
  for (int i = tid; i < BATCH * NC; i += TPB) w_state[i] = 0.f;
  for (int i = tid; i < BATCH * NA; i += TPB) k_lds[i] = 0.f;
  for (int i = tid; i < GSL; i += TPB) {
    int gr = (i >> 5) * NH + wg * 32 + (i & 31);
    bias_lds[i] = b_ih[gr] + b_hh[gr];
  }
  async_wait0();

  float cell0[8], cell1[8];
#pragma unroll
  for (int r = 0; r < 8; ++r) { cell0[r] = 0.f; cell1[r] = 0.f; }
  float lossA = 0.f, maskA = 0.f;

  for (int t = 0; t < T_STEPS; ++t) {
    const int p = t & 1, pp = p ^ 1;
    __bf16* hcur = hglob + (size_t)p * BATCH * NH;
    const __bf16* hprev = hglob + (size_t)pp * BATCH * NH;

    // ---- (a) refresh stroke-point columns of x ----
    if (tid < BATCH) {
      const float* sp = seq_pt + ((size_t)t * BATCH + tid) * 3;
      xbuf[tid * XD + 0] = (__bf16)sp[0];
      xbuf[tid * XD + 1] = (__bf16)sp[1];
      xbuf[tid * XD + 2] = (__bf16)sp[2];
      if (t + 1 < T_STEPS)
        __builtin_prefetch(seq_pt + ((size_t)(t + 1) * BATCH + tid) * 3, 0, 1);
    }
    __syncthreads();

    // ---- (b) gates = x@Wih^T + h@Whh^T + bias via WMMA bf16 ----
    // wave = batch row-tile (16 rows), 8 col-tiles = [i|i|f|f|g|g|o|o]
    v8f acc[8];
#pragma unroll
    for (int ct = 0; ct < 8; ++ct) {
      float bb = bias_lds[16 * ct + nn];
#pragma unroll
      for (int r = 0; r < 8; ++r) acc[ct][r] = bb;
    }
#pragma unroll
    for (int kc = 0; kc < 3; ++kc) {            // x part (K = 96), LDS operands
      v16bf a = load_a16(xbuf + (16 * wave) * XD + kc * 32, XD, lane);
#pragma unroll
      for (int ct = 0; ct < 8; ++ct) {
        v16bf bt = load_b16(wih_lds + (16 * ct) * XD + kc * 32, XD, lane);
        acc[ct] = __builtin_amdgcn_wmma_f32_16x16x32_bf16(false, a, false, bt,
                                                          (short)0, acc[ct], false, false);
      }
    }
    for (int kc = 0; kc < 16; ++kc) {           // h part (K = 512), Whh in LDS
      v16bf a = load_a16(hprev + (size_t)(16 * wave) * NH + kc * 32, NH, lane);
#pragma unroll
      for (int ct = 0; ct < 8; ++ct) {
        v16bf bt = load_b16(whh_lds + (16 * ct) * NH + kc * 32, NH, lane);
        acc[ct] = __builtin_amdgcn_wmma_f32_16x16x32_bf16(false, a, false, bt,
                                                          (short)0, acc[ct], false, false);
      }
    }

    // ---- (c) LSTM pointwise; cell state stays in VGPRs; write h slice ----
#pragma unroll
    for (int r = 0; r < 8; ++r) {
      const int brow = 16 * wave + r + 8 * g;   // D-matrix: M = r + 8*(lane>>4)
      {
        float iv = sigf(acc[0][r]), fv = sigf(acc[2][r]);
        float gv = tanhf(acc[4][r]), ov = sigf(acc[6][r]);
        float cv = fv * cell0[r] + iv * gv; cell0[r] = cv;
        hcur[(size_t)brow * NH + wg * 32 + nn] = (__bf16)(ov * tanhf(cv));
      }
      {
        float iv = sigf(acc[1][r]), fv = sigf(acc[3][r]);
        float gv = tanhf(acc[5][r]), ov = sigf(acc[7][r]);
        float cv = fv * cell1[r] + iv * gv; cell1[r] = cv;
        hcur[(size_t)brow * NH + wg * 32 + 16 + nn] = (__bf16)(ov * tanhf(cv));
      }
    }

    // ---- grid barrier: h(t) fully written before anyone reads it ----
    __builtin_amdgcn_fence(__ATOMIC_RELEASE, "agent");
    __syncthreads();
    if (tid < 32) {                             // one wave per WG
      if (lane == 0) {
        atomicAdd(bar, 1);
        const int target = (t + 1) * NWG;       // monotonic, replay-safe
        while (__hip_atomic_load(bar, __ATOMIC_RELAXED, __HIP_MEMORY_SCOPE_AGENT) < target)
          __builtin_amdgcn_s_sleep(1);
      }
      __builtin_amdgcn_s_cluster_barrier();     // NOP unless dispatched as cluster
    }
    __syncthreads();
    __builtin_amdgcn_fence(__ATOMIC_ACQUIRE, "agent");

    // ---- (d) attention projection att = h @ Watt^T via WMMA (30 -> 32) ----
    {
      v8f acc2[2];
#pragma unroll
      for (int ct = 0; ct < 2; ++ct) {
        const int col = 16 * ct + nn;
        float bb = (col < 30) ? b_att[col] : 0.f;
#pragma unroll
        for (int r = 0; r < 8; ++r) acc2[ct][r] = bb;
      }
      for (int kc = 0; kc < 16; ++kc) {
        v16bf a = load_a16(hcur + (size_t)(16 * wave) * NH + kc * 32, NH, lane);
#pragma unroll
        for (int ct = 0; ct < 2; ++ct) {
          v16bf bt = load_b16(watt_lds + (16 * ct) * NH + kc * 32, NH, lane);
          acc2[ct] = __builtin_amdgcn_wmma_f32_16x16x32_bf16(false, a, false, bt,
                                                             (short)0, acc2[ct], false, false);
        }
      }
#pragma unroll
      for (int ct = 0; ct < 2; ++ct)
#pragma unroll
        for (int r = 0; r < 8; ++r)
          att_lds[(16 * wave + r + 8 * g) * 32 + 16 * ct + nn] = acc2[ct][r];
    }
    __syncthreads();

    // ---- (e) attention window state update ----
    if (tid < BATCH) {                          // alpha/beta/k_new + k,w masking
      const int b = tid;
      const float m = seq_mask[(size_t)t * BATCH + b];
#pragma unroll
      for (int a = 0; a < NA; ++a) {
        float ah = att_lds[b * 32 + a], bh = att_lds[b * 32 + 10 + a], kh = att_lds[b * 32 + 20 + a];
        float kold = k_lds[b * NA + a];
        float knew = kold + expf(kh);
        att_lds[b * 32 + a]      = expf(ah);    // alpha
        att_lds[b * 32 + 10 + a] = expf(bh);    // beta
        att_lds[b * 32 + 20 + a] = knew;        // k_new
        k_lds[b * NA + a] = m * knew + (1.f - m) * kold;
      }
      const float om = 1.f - m;
      for (int ch = 0; ch < NC; ++ch) w_state[b * NC + ch] *= om;
    }
    __syncthreads();
    {                                           // phi + scatter into w (2 half-u threads per b)
      const int b = tid & (BATCH - 1), uh = tid >> 7;
      const float m = seq_mask[(size_t)t * BATCH + b];
      float al[NA], be[NA], kn[NA];
#pragma unroll
      for (int a = 0; a < NA; ++a) {
        al[a] = att_lds[b * 32 + a];
        be[a] = att_lds[b * 32 + 10 + a];
        kn[a] = att_lds[b * 32 + 20 + a];
      }
      for (int u = uh * 32; u < uh * 32 + 32; ++u) {
        float phi = 0.f;
#pragma unroll
        for (int a = 0; a < NA; ++a) { float d = kn[a] - (float)u; phi += al[a] * expf(-be[a] * d * d); }
        phi *= m * c_mask[(size_t)u * BATCH + b];
        atomicAdd(&w_state[b * NC + cidx[(size_t)u * BATCH + b]], phi);  // ds_add_f32
      }
    }
    __syncthreads();
    if (tid < BATCH) {                          // publish w(t) into x for step t+1
      for (int ch = 0; ch < NC; ++ch)
        xbuf[tid * XD + 3 + ch] = (__bf16)w_state[tid * NC + ch];
    }
    __syncthreads();

    // ---- (f) mixture head + NLL (WGs 0..7, 16 batches each) ----
    if (wg < 8) {
      // h rows -> hwbuf via async global->LDS (ASYNCcnt); w + pad via LDS copy
      for (int idx = tid; idx < 16 * (NH / 8); idx += TPB) {      // 1024 x 16B chunks
        const int r = idx / (NH / 8), cseg = idx % (NH / 8);
        const __bf16* src = hcur + (size_t)(16 * wg + r) * NH + cseg * 8;
        async_copy_b128(lds_addr_of(hwbuf + r * MIXIN + cseg * 8), src);
      }
      for (int idx = tid; idx < 16 * (MIXIN - NH); idx += TPB) {
        const int r = idx / (MIXIN - NH), cc = idx % (MIXIN - NH);
        hwbuf[r * MIXIN + NH + cc] =
            (cc < NC) ? xbuf[(16 * wg + r) * XD + 3 + cc] : (__bf16)0.f;
      }
      async_wait0();
      __syncthreads();
      {
        const int col = 16 * wave + nn;         // wave owns one output col-tile
        v8f my;
        float bb = (col < 121) ? b_mix[col] : 0.f;
#pragma unroll
        for (int r = 0; r < 8; ++r) my[r] = bb;
        for (int kc = 0; kc < 19; ++kc) {
          v16bf a  = load_a16(hwbuf + kc * 32, MIXIN, lane);
          v16bf bt = load_b16(wmix + (size_t)(16 * wave) * MIXIN + kc * 32, MIXIN, lane);
          my = __builtin_amdgcn_wmma_f32_16x16x32_bf16(false, a, false, bt,
                                                       (short)0, my, false, false);
        }
#pragma unroll
        for (int r = 0; r < 8; ++r) y_lds[(r + 8 * g) * MIXO + col] = my[r];
      }
      __syncthreads();
      if (tid < 16) {
        const int bl = tid, b = 16 * wg + bl;
        const float* y = y_lds + bl * MIXO;
        const float m   = seq_mask[(size_t)t * BATCH + b];
        const float x1  = tgt[((size_t)t * BATCH + b) * 3 + 0];
        const float x2  = tgt[((size_t)t * BATCH + b) * 3 + 1];
        const float pen = tgt[((size_t)t * BATCH + b) * 3 + 2];
        float mA = -1e30f, mB = -1e30f;
        for (int j2 = 0; j2 < NG; ++j2) {
          float pi = y[j2];
          float ln = mix_logn(y, j2, x1, x2);
          mA = fmaxf(mA, pi); mB = fmaxf(mB, pi + ln);
        }
        float sA = 0.f, sB = 0.f;
        for (int j2 = 0; j2 < NG; ++j2) {
          sA += expf(y[j2] - mA);
          sB += expf(y[j2] + mix_logn(y, j2, x1, x2) - mB);
        }
        float nllg = (mA + logf(sA)) - (mB + logf(sB));   // lse(pi) - lse(pi+logn)
        float z = y[120];
        float nllp = -(pen * logsigf(z) + (1.f - pen) * logsigf(-z));
        lossA += m * (nllg + nllp);
        maskA += m;
      }
    }
  } // t loop

  // ---- final reduction into global accumulators ----
  __syncthreads();
  red[tid] = lossA; __syncthreads();
  for (int s = TPB / 2; s > 0; s >>= 1) { if (tid < s) red[tid] += red[tid + s]; __syncthreads(); }
  if (tid == 0) atomicAdd(&accum[0], red[0]);
  __syncthreads();
  red[tid] = maskA; __syncthreads();
  for (int s = TPB / 2; s > 0; s >>= 1) { if (tid < s) red[tid] += red[tid + s]; __syncthreads(); }
  if (tid == 0) atomicAdd(&accum[1], red[0]);
}

__global__ void hw_finalize(const char* __restrict__ ws, float* __restrict__ out) {
  const float* accum = (const float*)(ws + OFF_HDR + 64);
  out[0] = accum[0] / accum[1];
}

// ---------------------------------------------------------------------------
extern "C" void kernel_launch(void* const* d_in, const int* in_sizes, int n_in,
                              void* d_out, int out_size, void* d_ws, size_t ws_size,
                              hipStream_t stream) {
  const float* seq_pt   = (const float*)d_in[0];
  const float* seq_mask = (const float*)d_in[1];
  const float* tgt      = (const float*)d_in[2];
  const int*   c        = (const int*)d_in[3];
  const float* c_mask   = (const float*)d_in[4];
  const float* W_ih     = (const float*)d_in[5];
  const float* b_ih     = (const float*)d_in[6];
  const float* W_hh     = (const float*)d_in[7];
  const float* b_hh     = (const float*)d_in[8];
  const float* W_att    = (const float*)d_in[9];
  const float* b_att    = (const float*)d_in[10];
  const float* W_mix    = (const float*)d_in[11];
  const float* b_mix    = (const float*)d_in[12];
  char* ws = (char*)d_ws;

  hw_prep<<<512, TPB, 0, stream>>>(W_ih, W_hh, W_att, W_mix, ws);
  hw_main<<<NWG, TPB, SMEM_BYTES, stream>>>(seq_pt, seq_mask, tgt, c, c_mask,
                                            b_ih, b_hh, b_att, b_mix, ws);
  hw_finalize<<<1, 1, 0, stream>>>(ws, (float*)d_out);
}
